// CausalWordPropagation_3255585210485
// MI455X (gfx1250) — matrix operand order repaired and checked by reading, not compiled, not run on hardware
//
#include <hip/hip_runtime.h>
#include <stdint.h>

// ---- problem constants (match reference) ----
#define NB 4
#define NT 4096
#define NV 1024
#define TT 32           // t rows per block (2 subtiles of 16)
#define SB 128          // s-block per outer iteration
#define SPITCH 130      // padded pitch of f32 S tile in LDS (floats, fallback path)
#define SP2 136         // padded pitch of bf16 S tile in LDS (shorts, fast path)
#define QP 1048         // padded pitch of Q tile in LDS (bf16 elems; 12*ln%64 bank perm)

typedef __attribute__((ext_vector_type(16))) __bf16 v16bf;
typedef __attribute__((ext_vector_type(8)))  float  v8f;
typedef __attribute__((ext_vector_type(2)))  float  v2f;

union BF16x16 {
  v16bf v;
  __bf16 e[16];
  uint4 q[2];
};
union BF16x4 {
  ushort4 u4;
  __bf16  e[4];
};

// ---- pre-pass: one LDS-tiled sweep produces BOTH bf16 copies of x:
//   xbf[b][s][v] (row-major, phase-1 B operand)
//   xT [b][v][s] (transposed, phase-2 B operand -> s-major contiguous loads)
__global__ __launch_bounds__(256)
void prep_bf16(const float* __restrict__ x,
               unsigned short* __restrict__ xbf,
               unsigned short* __restrict__ xT)
{
  __shared__ unsigned short tile[64][66];   // 64s x 64v tile, padded
  const int nv64 = NV / 64;
  const int b  = blockIdx.x / ((NT / 64) * nv64);
  const int r  = blockIdx.x % ((NT / 64) * nv64);
  const int s0 = (r / nv64) * 64;
  const int v0 = (r % nv64) * 64;
  const int tid = threadIdx.x;
  const int tv = (tid & 15) * 4;
  const int ts = tid >> 4;

  const float* xb = x + (size_t)b * NT * NV;
  unsigned short* xbf_b = xbf + (size_t)b * NT * NV;
  #pragma unroll
  for (int p = 0; p < 4; ++p) {
    const int s = ts + p * 16;
    const float4 f = *(const float4*)(xb + (size_t)(s0 + s) * NV + v0 + tv);
    BF16x4 o;
    o.e[0] = (__bf16)f.x; o.e[1] = (__bf16)f.y;
    o.e[2] = (__bf16)f.z; o.e[3] = (__bf16)f.w;
    *(ushort4*)(xbf_b + (size_t)(s0 + s) * NV + v0 + tv) = o.u4;   // coalesced
    tile[s][tv + 0] = o.u4.x; tile[s][tv + 1] = o.u4.y;
    tile[s][tv + 2] = o.u4.z; tile[s][tv + 3] = o.u4.w;
  }
  __syncthreads();
  unsigned short* xT_b = xT + (size_t)b * NV * NT;
  #pragma unroll
  for (int p = 0; p < 4; ++p) {
    const int v = ts + p * 16;
    ushort4 o;
    o.x = tile[tv + 0][v]; o.y = tile[tv + 1][v];
    o.z = tile[tv + 2][v]; o.w = tile[tv + 3][v];
    *(ushort4*)(xT_b + (size_t)(v0 + v) * NT + s0 + tv) = o;       // coalesced
  }
}

__global__ __launch_bounds__(512)
void cwp_fused(const float* __restrict__ x,
               const unsigned short* __restrict__ xbf,  // bf16 x, or null
               const unsigned short* __restrict__ xT,   // bf16 x^T, or null
               const float* __restrict__ decay_logit,
               const float* __restrict__ out_scale_p,
               const float* __restrict__ q_scale,
               const float* __restrict__ k_scale,
               float* __restrict__ out)
{
  __shared__ alignas(16) __bf16 Qbf[TT * QP];              // q*k-scaled query rows
  __shared__ alignas(16) char   Sbuf[TT * SPITCH * 4];     // overlaid S tile
  __bf16* S16  = (__bf16*)Sbuf;                            // fast: TT x SP2 bf16
  float*  Sldf = (float*)Sbuf;                             // fallback: TT x SPITCH f32

  const int tid  = threadIdx.x;
  const int W    = tid >> 5;     // wave 0..15
  const int w    = W >> 1;       // v-slice 0..7
  const int tsub = W & 1;        // t-subtile 0..1
  const int toff = 16 * tsub;
  const int lane = tid & 31;
  const int ln   = lane & 15;
  const int lh   = lane >> 4;

  const int b     = blockIdx.x / (NT / TT);
  const int tbase = (blockIdx.x % (NT / TT)) * TT;

  const float* xb = x + (size_t)b * NT * NV;
  const unsigned short* xbf_b = xbf ? (xbf + (size_t)b * NT * NV) : nullptr;
  const unsigned short* xT_b  = xT  ? (xT  + (size_t)b * NV * NT) : nullptr;
  const bool fast = (xbf_b != nullptr) && (xT_b != nullptr);   // block-uniform

  // ---- stage Q tile (folded q_scale*k_scale) into LDS as bf16 ----
  for (int e = tid; e < TT * NV; e += 512) {
    const int row = e >> 10, col = e & (NV - 1);
    const float v = xb[(size_t)(tbase + row) * NV + col] * q_scale[col] * k_scale[col];
    Qbf[row * QP + col] = (__bf16)v;
  }
  __syncthreads();

  const float decay     = 1.0f / (1.0f + __expf(-decay_logit[0]));
  const float log_decay = __logf(decay);
  const float oscale    = out_scale_p[0];

  // w(t,s) = decay^(s-tbase-1) * decay^-(t-tbase), t-tbase = toff + r + 8*lh
  float etv[8];
  #pragma unroll
  for (int r = 0; r < 8; ++r)
    etv[r] = __expf(-log_decay * (float)(toff + r + 8 * lh));

  v8f acc[8] = {};               // 16t x 128v accumulator (this wave's tile)
  const int vw = w * 128;

  const int sb_start = (tbase / SB) * SB;    // weight mask kills s <= t
  for (int sb = sb_start; sb < NT; sb += SB) {
    // ==== Phase 1: wave computes S subtile (tsub, s in [sb+16w, +16)) ====
    const int srow = sb + 16 * w + ln;       // this lane's s column
    v8f sacc = {};
    if (fast) {
      if (sb + SB < NT)
        __builtin_prefetch(xbf_b + (size_t)(srow + SB) * NV, 0, 1);
      const uint4* brow16 = (const uint4*)(xbf_b + (size_t)srow * NV);
      for (int k = 0; k < NV; k += 32) {
        BF16x16 A, Bv;
        A.q[0] = *(const uint4*)&Qbf[(toff + ln) * QP + k + 8 * lh];
        A.q[1] = *(const uint4*)&Qbf[(toff + ln) * QP + k + 16 + 8 * lh];
        const int u = (k + 8 * lh) >> 3;     // K(=v) = k+8*lh+{0..7}, k+16+8*lh+{0..7}
        Bv.q[0] = brow16[u];
        Bv.q[1] = brow16[u + 2];
        sacc = __builtin_amdgcn_wmma_f32_16x16x32_bf16(
            false, A.v, false, Bv.v, (short)0, sacc, false, false);
      }
    } else {
      const float* brow = xb + (size_t)srow * NV;
      for (int k = 0; k < NV; k += 32) {
        BF16x16 A, Bv;
        A.q[0] = *(const uint4*)&Qbf[(toff + ln) * QP + k + 8 * lh];
        A.q[1] = *(const uint4*)&Qbf[(toff + ln) * QP + k + 16 + 8 * lh];
        alignas(16) float bw[16];
        ((float4*)bw)[0] = ((const float4*)(brow + k + 8 * lh))[0];
        ((float4*)bw)[1] = ((const float4*)(brow + k + 8 * lh))[1];
        ((float4*)bw)[2] = ((const float4*)(brow + k + 16 + 8 * lh))[0];
        ((float4*)bw)[3] = ((const float4*)(brow + k + 16 + 8 * lh))[1];
        #pragma unroll
        for (int i = 0; i < 16; ++i) Bv.e[i] = (__bf16)bw[i];
        sacc = __builtin_amdgcn_wmma_f32_16x16x32_bf16(
            false, A.v, false, Bv.v, (short)0, sacc, false, false);
      }
    }

    // decay weights (1 exp per s-block); D layout: VGPR r -> M = r+8*lh, N = ln
    const int   dt = srow - tbase;
    const float es = __expf(log_decay * (float)(dt - 1));

    if (fast) {
      // ==== fast Phase 2: bf16 WMMA, S from LDS (A), x^T from global (B) ====
      #pragma unroll
      for (int r = 0; r < 8; ++r) {
        const float wgt = (dt > toff + r + 8 * lh) ? es * etv[r] : 0.0f;
        S16[(toff + r + 8 * lh) * SP2 + 16 * w + ln] = (__bf16)(sacc[r] * wgt);
      }
      __syncthreads();
      for (int k = 0; k < SB; k += 32) {
        BF16x16 A;  // S tile 16t x 32s: lane row t, K(=s) chunks
        A.q[0] = *(const uint4*)&S16[(toff + ln) * SP2 + k + 8 * lh];
        A.q[1] = *(const uint4*)&S16[(toff + ln) * SP2 + k + 16 + 8 * lh];
        #pragma unroll
        for (int j = 0; j < 8; ++j) {
          const int vcol = vw + 16 * j + ln;
          const uint4* bp = (const uint4*)(xT_b + (size_t)vcol * NT + sb + k + 8 * lh);
          BF16x16 Bv;  // X 32s x 16v: lane col v, K(=s) chunks k+8*lh, k+16+8*lh
          Bv.q[0] = bp[0];
          Bv.q[1] = bp[2];
          acc[j] = __builtin_amdgcn_wmma_f32_16x16x32_bf16(
              false, A.v, false, Bv.v, (short)0, acc[j], false, false);
        }
      }
    } else {
      // ==== fallback Phase 2: f32 WMMA (reference precision) ====
      #pragma unroll
      for (int r = 0; r < 8; ++r) {
        const float wgt = (dt > toff + r + 8 * lh) ? es * etv[r] : 0.0f;
        Sldf[(toff + r + 8 * lh) * SPITCH + 16 * w + ln] = sacc[r] * wgt;
      }
      __syncthreads();
      for (int k = 0; k < SB; k += 4) {
        const float2 af = *(const float2*)&Sldf[(toff + ln) * SPITCH + k + 2 * lh];
        v2f av; av.x = af.x; av.y = af.y;
        const float* bbase = xb + (size_t)(sb + k + 2 * lh) * NV;
        #pragma unroll
        for (int j = 0; j < 8; ++j) {
          const int vcol = vw + 16 * j + ln;
          v2f bv;
          bv.x = bbase[vcol];
          bv.y = bbase[NV + vcol];
          acc[j] = __builtin_amdgcn_wmma_f32_16x16x4_f32(
              false, av, false, bv, (short)0, acc[j], false, false);
        }
      }
    }
    __syncthreads();
  }

  // ---- store: D layout VGPR r -> t = tbase + toff + r + 8*lh ----
  float* ob = out + (size_t)b * NT * NV;
  #pragma unroll
  for (int j = 0; j < 8; ++j) {
    const int vcol = vw + 16 * j + ln;
    #pragma unroll
    for (int r = 0; r < 8; ++r) {
      const int t = tbase + toff + r + 8 * lh;
      ob[(size_t)t * NV + vcol] = acc[j][r] * oscale;
    }
  }
}

extern "C" void kernel_launch(void* const* d_in, const int* in_sizes, int n_in,
                              void* d_out, int out_size, void* d_ws, size_t ws_size,
                              hipStream_t stream) {
  (void)in_sizes; (void)n_in; (void)out_size;
  const float* x  = (const float*)d_in[0];
  const float* dl = (const float*)d_in[1];
  const float* os = (const float*)d_in[2];
  const float* qs = (const float*)d_in[3];
  const float* ks = (const float*)d_in[4];
  float* out = (float*)d_out;

  const size_t half = (size_t)NB * NT * NV * sizeof(unsigned short);  // 32 MiB
  unsigned short* xbf = nullptr;
  unsigned short* xT  = nullptr;
  if (d_ws && ws_size >= 2 * half) {
    xbf = (unsigned short*)d_ws;
    xT  = (unsigned short*)((char*)d_ws + half);
    const int nblk = NB * (NT / 64) * (NV / 64);   // 4096 tiles of 64x64
    hipLaunchKernelGGL(prep_bf16, dim3(nblk), dim3(256), 0, stream, x, xbf, xT);
  }

  dim3 grid(NB * (NT / TT));
  dim3 block(512);
  hipLaunchKernelGGL(cwp_fused, grid, block, 0, stream,
                     x, xbf, xT, dl, os, qs, ks, out);
}
